// LinearAttention_51883204935736
// MI455X (gfx1250) — matrix operand order repaired and checked by reading, not compiled, not run on hardware
//
#include <hip/hip_runtime.h>
#include <hip/hip_bf16.h>

// CDNA5 / gfx1250 linear-attention pipeline, bf16 WMMA with fp32 accumulate.
// Matrix math: v_wmma_f32_16x16x32_bf16. Pure-copy LDS staging uses
// global_load_async_to_lds_b128 (ASYNCcnt) instead of load+ds_store.

typedef __bf16 bf16_t;
typedef __attribute__((ext_vector_type(16))) __bf16 v16bf;
typedef __attribute__((ext_vector_type(8)))  __bf16 v8bf;
typedef __attribute__((ext_vector_type(8)))  float  v8f;

union Frag {
    v16bf v;
    v8bf  h[2];
};

__device__ __forceinline__ v8f wmma_bf16(const Frag& a, const Frag& b, v8f c) {
    // (neg_a, A, neg_b, B, c_mod, C, reuse_a, reuse_b)
    return __builtin_amdgcn_wmma_f32_16x16x32_bf16(false, a.v, false, b.v,
                                                   (short)0, c, false, false);
}

__device__ __forceinline__ v8f zero8() {
    v8f z;
#pragma unroll
    for (int e = 0; e < 8; ++e) z[e] = 0.0f;
    return z;
}

// 16 bytes/lane global -> LDS, tracked by ASYNCcnt.
// Generic LDS pointer's low 32 bits are the LDS byte offset.
__device__ __forceinline__ void async_copy_b128(void* lds, const void* gaddr) {
    unsigned lds_off = (unsigned)(unsigned long long)lds;
    asm volatile("global_load_async_to_lds_b128 %0, %1, off"
                 :: "v"(lds_off), "v"(gaddr)
                 : "memory");
}

__device__ __forceinline__ void wait_async0() {
#if __has_builtin(__builtin_amdgcn_s_wait_asynccnt)
    __builtin_amdgcn_s_wait_asynccnt(0);
#else
    asm volatile("s_wait_asynccnt 0" ::: "memory");
#endif
}

// ---------------------------------------------------------------------------
// Generic GEMM: Out[M,N] = act(A[M,K] @ W[K,N]) (+bias)
// AIN_F32: A is f32 (convert to bf16 during LDS staging) else bf16 (async DMA).
// ACT: 1 -> elu(x)+1 feature map. OUT_F32: f32 store (+bias) else bf16 store.
// Block tile 128x128, K-step 32, 256 threads = 8 waves, 2x4 WMMA tiles/wave.
// ---------------------------------------------------------------------------
template <int AIN_F32, int ACT, int OUT_F32>
__global__ __launch_bounds__(256) void gemm_wmma_kernel(
    const void* __restrict__ Ain, const float* __restrict__ W,
    const float* __restrict__ bias, void* __restrict__ Out,
    int M, int K, int N)
{
    __shared__ __attribute__((aligned(16))) bf16_t As[128][40];  // [m][k], padded
    __shared__ __attribute__((aligned(16))) bf16_t Bs[128][40];  // [n][k], padded

    const int tid  = threadIdx.x;
    const int lane = tid & 31;
    const int wv   = tid >> 5;
    const int wm   = (wv & 3) * 32;   // wave m-offset in block
    const int wn   = (wv >> 2) * 64;  // wave n-offset in block
    const int lm   = lane & 15;
    const int kh   = lane >> 4;

    const int nblk = blockIdx.x * 128;
    const int mblk = blockIdx.y * 128;

    v8f acc[2][4];
#pragma unroll
    for (int i = 0; i < 2; ++i)
#pragma unroll
        for (int j = 0; j < 4; ++j) acc[i][j] = zero8();

    for (int kb = 0; kb < K; kb += 32) {
        __syncthreads();
        // ---- stage A tile (128x32) as bf16 ----
        if (AIN_F32) {
            const float* A = (const float*)Ain;
#pragma unroll
            for (int i = 0; i < 4; ++i) {
                int f   = tid + i * 256;      // float4 id, 8 per row
                int row = f >> 3;
                int cq  = f & 7;
                const float4 x =
                    *(const float4*)(A + (size_t)(mblk + row) * K + kb + cq * 4);
                bf16_t* dst = &As[row][cq * 4];
                dst[0] = (bf16_t)x.x; dst[1] = (bf16_t)x.y;
                dst[2] = (bf16_t)x.z; dst[3] = (bf16_t)x.w;
            }
        } else {
            const bf16_t* A = (const bf16_t*)Ain;
#pragma unroll
            for (int i = 0; i < 2; ++i) {
                int c   = tid + i * 256;      // 8-elem (16B) chunk id, 4 per row
                int row = c >> 2;
                int cq  = c & 3;
                async_copy_b128(&As[row][cq * 8],
                                A + (size_t)(mblk + row) * K + kb + cq * 8);
            }
        }
        // ---- stage W tile (32x128) transposed -> Bs[n][k], f32->bf16 ----
#pragma unroll
        for (int i = 0; i < 4; ++i) {
            int f  = tid + i * 256;           // float4 id, 32 per k-row
            int kr = f >> 5;
            int cq = f & 31;
            const float4 x =
                *(const float4*)(W + (size_t)(kb + kr) * N + nblk + cq * 4);
            Bs[cq * 4 + 0][kr] = (bf16_t)x.x;
            Bs[cq * 4 + 1][kr] = (bf16_t)x.y;
            Bs[cq * 4 + 2][kr] = (bf16_t)x.z;
            Bs[cq * 4 + 3][kr] = (bf16_t)x.w;
        }
        if (!AIN_F32) wait_async0();
        __syncthreads();

        // ---- fragments + WMMA ----
        Frag a[2], b[4];
#pragma unroll
        for (int i = 0; i < 2; ++i) {
            const int r = wm + i * 16 + lm;
            a[i].h[0] = *(const v8bf*)&As[r][kh * 8];        // K = 8kh..8kh+7
            a[i].h[1] = *(const v8bf*)&As[r][16 + kh * 8];   // K = 16+8kh..
        }
#pragma unroll
        for (int j = 0; j < 4; ++j) {
            const int c = wn + j * 16 + lm;
            b[j].h[0] = *(const v8bf*)&Bs[c][kh * 8];        // K = 8kh..8kh+7
            b[j].h[1] = *(const v8bf*)&Bs[c][16 + kh * 8];   // K = 16+8kh..
        }
#pragma unroll
        for (int i = 0; i < 2; ++i)
#pragma unroll
            for (int j = 0; j < 4; ++j)
                acc[i][j] = wmma_bf16(a[i], b[j], acc[i][j]);
    }

    // ---- epilogue ----
#pragma unroll
    for (int i = 0; i < 2; ++i) {
#pragma unroll
        for (int j = 0; j < 4; ++j) {
#pragma unroll
            for (int e = 0; e < 8; ++e) {
                float val = acc[i][j][e];
                if (ACT) val = (val > 0.0f) ? (val + 1.0f) : __expf(val);
                const int row = mblk + wm + i * 16 + kh * 8 + e;
                const int col = nblk + wn + j * 16 + lm;
                if (OUT_F32) {
                    float bv = bias ? bias[col] : 0.0f;
                    ((float*)Out)[(size_t)row * N + col] = val + bv;
                } else {
                    ((bf16_t*)Out)[(size_t)row * N + col] = (bf16_t)val;
                }
            }
        }
    }
}

// ---------------------------------------------------------------------------
// kv[b,h] = sum_t kf[t,:]^T vh[t,:]  (64x64, K=T=4096), ksum[b,h] = sum_t kf + EPS
// One block per (b,h). 8 waves, each owns one 16-row m-tile x two 16-col n-tiles.
// ---------------------------------------------------------------------------
__global__ __launch_bounds__(256) void kv_ksum_kernel(
    const bf16_t* __restrict__ kf, const bf16_t* __restrict__ vh,
    float* __restrict__ kv, float* __restrict__ ksum)
{
    __shared__ __attribute__((aligned(16))) bf16_t kT[64][40];  // [d][t]
    __shared__ __attribute__((aligned(16))) bf16_t vT[64][40];  // [e][t]

    const int bh = blockIdx.x;
    const int b  = bh >> 4, h = bh & 15;
    const bf16_t* kbp = kf + (size_t)b * 4096 * 1024 + h * 64;
    const bf16_t* vbp = vh + (size_t)b * 4096 * 1024 + h * 64;

    const int tid  = threadIdx.x;
    const int lane = tid & 31;
    const int wv   = tid >> 5;
    const int lm   = lane & 15, kh = lane >> 4;
    const int mt   = (wv & 3) * 16;        // d-tile base
    const int nt   = (wv >> 2) * 32;       // e-tile base (2 tiles)

    v8f acc[2];
    acc[0] = zero8(); acc[1] = zero8();
    float ks = 0.0f;

    const int trow = tid >> 3;             // 0..31
    const int d0   = (tid & 7) * 8;        // 0..56

    for (int t0 = 0; t0 < 4096; t0 += 32) {
        __syncthreads();
        v8bf x = *(const v8bf*)(kbp + (size_t)(t0 + trow) * 1024 + d0);
        v8bf y = *(const v8bf*)(vbp + (size_t)(t0 + trow) * 1024 + d0);
#pragma unroll
        for (int j = 0; j < 8; ++j) {
            kT[d0 + j][trow] = x[j];
            vT[d0 + j][trow] = y[j];
        }
        __syncthreads();

        if (tid < 64) {
#pragma unroll
            for (int t = 0; t < 32; ++t) ks += (float)kT[tid][t];
        }

        Frag a;
        a.h[0] = *(const v8bf*)&kT[mt + lm][kh * 8];
        a.h[1] = *(const v8bf*)&kT[mt + lm][16 + kh * 8];
#pragma unroll
        for (int j = 0; j < 2; ++j) {
            Frag bb;
            const int cc = nt + j * 16 + lm;
            bb.h[0] = *(const v8bf*)&vT[cc][kh * 8];
            bb.h[1] = *(const v8bf*)&vT[cc][16 + kh * 8];
            acc[j] = wmma_bf16(a, bb, acc[j]);
        }
    }

    float* kvo = kv + (size_t)bh * 64 * 64;
#pragma unroll
    for (int j = 0; j < 2; ++j)
#pragma unroll
        for (int e = 0; e < 8; ++e) {
            const int d = mt + kh * 8 + e;
            const int c = nt + j * 16 + lm;
            kvo[d * 64 + c] = acc[j][e];
        }
    if (tid < 64) ksum[(size_t)bh * 64 + tid] = ks + 1e-6f;
}

// ---------------------------------------------------------------------------
// ao[t, h*64+e] = (qf[t,:] @ kv[h]) / (qf[t,:] . ksum[h])   for a 128-row tile
// grid = (T/128, H, B). qf tile staged via async DMA to LDS.
// ---------------------------------------------------------------------------
__global__ __launch_bounds__(256) void attn_apply_kernel(
    const bf16_t* __restrict__ qf, const float* __restrict__ kv,
    const float* __restrict__ ksum, bf16_t* __restrict__ ao)
{
    __shared__ __attribute__((aligned(16))) bf16_t qs[128][72];  // [t][d]
    __shared__ __attribute__((aligned(16))) bf16_t kvT[64][72];  // [e][d]
    __shared__ float den[128];
    __shared__ float ksl[64];

    const int tc = blockIdx.x;
    const int h  = blockIdx.y;
    const int b  = blockIdx.z;
    const int bh = b * 16 + h;

    const int tid  = threadIdx.x;
    const int lane = tid & 31;
    const int wv   = tid >> 5;
    const int lm   = lane & 15, kh = lane >> 4;

    const size_t grow0 = (size_t)b * 4096 + (size_t)tc * 128;
    const bf16_t* qb = qf + grow0 * 1024 + h * 64;

    // stage qf tile 128x64 (pure bf16 copy -> async DMA straight into LDS)
#pragma unroll
    for (int i = 0; i < 4; ++i) {
        int c = tid + i * 256;
        int row = c >> 3, dd = (c & 7) * 8;
        async_copy_b128(&qs[row][dd], qb + (size_t)row * 1024 + dd);
    }
    // stage kv (f32) transposed -> kvT[e][d] bf16
    const float* kvb = kv + (size_t)bh * 64 * 64;
#pragma unroll
    for (int i = 0; i < 4; ++i) {
        int f = tid + i * 256;
        int d = f >> 4, eq = f & 15;
        const float4 x = *(const float4*)(kvb + d * 64 + eq * 4);
        kvT[eq * 4 + 0][d] = (bf16_t)x.x;
        kvT[eq * 4 + 1][d] = (bf16_t)x.y;
        kvT[eq * 4 + 2][d] = (bf16_t)x.z;
        kvT[eq * 4 + 3][d] = (bf16_t)x.w;
    }
    if (tid < 64) ksl[tid] = ksum[(size_t)bh * 64 + tid];
    wait_async0();
    __syncthreads();

    if (tid < 128) {
        float s = 0.0f;
#pragma unroll
        for (int d = 0; d < 64; ++d) s += (float)qs[tid][d] * ksl[d];
        den[tid] = s;
    }
    __syncthreads();

    v8f acc[4];
#pragma unroll
    for (int j = 0; j < 4; ++j) acc[j] = zero8();

#pragma unroll
    for (int ksb = 0; ksb < 64; ksb += 32) {
        Frag a;
        a.h[0] = *(const v8bf*)&qs[wv * 16 + lm][ksb + kh * 8];
        a.h[1] = *(const v8bf*)&qs[wv * 16 + lm][ksb + 16 + kh * 8];
#pragma unroll
        for (int j = 0; j < 4; ++j) {
            Frag bb;
            bb.h[0] = *(const v8bf*)&kvT[j * 16 + lm][ksb + kh * 8];
            bb.h[1] = *(const v8bf*)&kvT[j * 16 + lm][ksb + 16 + kh * 8];
            acc[j] = wmma_bf16(a, bb, acc[j]);
        }
    }

    bf16_t* aob = ao + grow0 * 1024 + h * 64;
#pragma unroll
    for (int j = 0; j < 4; ++j)
#pragma unroll
        for (int e = 0; e < 8; ++e) {
            const int row = wv * 16 + kh * 8 + e;
            const int col = j * 16 + lm;
            aob[(size_t)row * 1024 + col] = (bf16_t)(acc[j][e] / den[row]);
        }
}

// ---------------------------------------------------------------------------
extern "C" void kernel_launch(void* const* d_in, const int* in_sizes, int n_in,
                              void* d_out, int out_size, void* d_ws, size_t ws_size,
                              hipStream_t stream) {
    (void)in_sizes; (void)n_in; (void)out_size; (void)ws_size;
    const float* v_in = (const float*)d_in[0];
    const float* k_in = (const float*)d_in[1];
    const float* q_in = (const float*)d_in[2];
    const float* Wq   = (const float*)d_in[3];
    const float* Wk   = (const float*)d_in[4];
    const float* Wv   = (const float*)d_in[5];
    const float* Wo   = (const float*)d_in[6];
    const float* bo   = (const float*)d_in[7];

    const size_t MT = (size_t)4 * 4096;  // 16384 rows
    bf16_t* qf = (bf16_t*)d_ws;
    bf16_t* kf = qf + MT * 1024;
    bf16_t* vh = kf + MT * 1024;
    float*  kvb  = (float*)(vh + MT * 1024);
    float*  ksum = kvb + 64 * 64 * 64;
    bf16_t* ao = kf;  // kf dead after kv_ksum_kernel; reuse for attn output

    dim3 blk(256);
    dim3 gg(1024 / 128, 16384 / 128);
    gemm_wmma_kernel<1, 1, 0><<<gg, blk, 0, stream>>>(q_in, Wq, nullptr, qf, 16384, 1024, 1024);
    gemm_wmma_kernel<1, 1, 0><<<gg, blk, 0, stream>>>(k_in, Wk, nullptr, kf, 16384, 1024, 1024);
    gemm_wmma_kernel<1, 0, 0><<<gg, blk, 0, stream>>>(v_in, Wv, nullptr, vh, 16384, 1024, 1024);
    kv_ksum_kernel<<<dim3(64), blk, 0, stream>>>(kf, vh, kvb, ksum);
    attn_apply_kernel<<<dim3(32, 16, 4), blk, 0, stream>>>(qf, kvb, ksum, ao);
    gemm_wmma_kernel<0, 0, 1><<<gg, blk, 0, stream>>>(ao, Wo, bo, d_out, 16384, 1024, 1024);
}
